// TransformerLayer_22058952032518
// MI455X (gfx1250) — matrix operand order repaired and checked by reading, not compiled, not run on hardware
//
#include <hip/hip_runtime.h>
#include <hip/hip_bf16.h>

typedef __attribute__((ext_vector_type(16))) _Float16 v16h;
typedef __attribute__((ext_vector_type(8)))  _Float16 v8h;
typedef __attribute__((ext_vector_type(4)))  _Float16 v4h;
typedef __attribute__((ext_vector_type(8)))  float    v8f;
typedef __attribute__((ext_vector_type(4)))  float    v4f;

// b128 beat type expected by the async-to-LDS builtin (per hipcc diagnostic).
typedef int v4i_b __attribute__((vector_size(16)));

#define LOG2E 1.4426950408889634f

#if defined(__AMDGCN__) && defined(__has_builtin)
#if __has_builtin(__builtin_amdgcn_global_load_async_to_lds_b128)
#define USE_ASYNC 1
#endif
#endif

#define AS1 __attribute__((address_space(1)))
#define AS3 __attribute__((address_space(3)))

// 16-byte global -> LDS copy; async (ASYNCcnt) when the gfx1250 builtin exists.
static __device__ __forceinline__ void copy8(const _Float16* g, _Float16* l) {
#ifdef USE_ASYNC
    __builtin_amdgcn_global_load_async_to_lds_b128(
        (AS1 v4i_b*)g, (AS3 v4i_b*)l, 0, 0);
#else
    *(v8h*)l = *(const v8h*)g;
#endif
}

static __device__ __forceinline__ void wait_async() {
#ifdef USE_ASYNC
#if __has_builtin(__builtin_amdgcn_s_wait_asynccnt)
    __builtin_amdgcn_s_wait_asynccnt(0);
#else
    asm volatile("s_wait_asynccnt 0x0" ::: "memory");
#endif
#endif
}

static __device__ __forceinline__ v16h frag_from(const _Float16* p0, const _Float16* p1) {
    union { v16h v; v8h h[2]; } u;
    u.h[0] = *(const v8h*)p0;
    u.h[1] = *(const v8h*)p1;
    return u.v;
}

static __device__ __forceinline__ v8f wmma_f16(v16h a, v16h b, v8f c) {
    return __builtin_amdgcn_wmma_f32_16x16x32_f16(false, a, false, b, (short)0, c, false, false);
}

// ---------------------------------------------------------------------------
// f32[R][C] -> f16[C][R] transposed conversion (weights become N-major so the
// GEMM B tile is a contiguous copy -> async-to-LDS capable).
// ---------------------------------------------------------------------------
__global__ void __launch_bounds__(256) cvt_f32_to_f16_t(const float* __restrict__ in,
                                                        _Float16* __restrict__ out,
                                                        int R, int C) {
    int i = blockIdx.x * 256 + threadIdx.x;      // over R*C, out-coalesced
    int c = i / R, r = i % R;
    out[i] = (_Float16)in[(size_t)r * C + c];
}

// ---------------------------------------------------------------------------
// Fused W_QKV f16, transposed: out[n][d], n in 0..3071 (Q|K|V, head-major 64).
// Sources [16][1024][64] f32.
// ---------------------------------------------------------------------------
__global__ void __launch_bounds__(256) build_wqkv_t(const float* __restrict__ wq,
                                                    const float* __restrict__ wk,
                                                    const float* __restrict__ wv,
                                                    _Float16* __restrict__ out) {
    int i = blockIdx.x * 256 + threadIdx.x;      // over 3072*1024
    int n = i >> 10, d = i & 1023;
    int part = n >> 10, within = n & 1023;
    int head = within >> 6, dim = within & 63;
    const float* src = (part == 0) ? wq : (part == 1) ? wk : wv;
    out[i] = (_Float16)src[((size_t)head * 1024 + d) * 64 + dim];
}

// ---------------------------------------------------------------------------
// Scatter qkv [4096][3072] -> q/k [b*h][s][64], v transposed [b*h][64][s].
// Q scaled by 1/sqrt(64).
// ---------------------------------------------------------------------------
__global__ void __launch_bounds__(256) scatter_qkv(const _Float16* __restrict__ qkv,
                                                   _Float16* __restrict__ q,
                                                   _Float16* __restrict__ k,
                                                   _Float16* __restrict__ vt) {
    int i = blockIdx.x * 256 + threadIdx.x;      // over 4096*3072
    int m = i / 3072, n = i % 3072;
    int part = n >> 10, within = n & 1023;
    int head = within >> 6, dim = within & 63;
    int b = m >> 11, s = m & 2047;
    int bh = b * 16 + head;
    _Float16 val = qkv[i];
    if (part == 0)
        q[((size_t)bh * 2048 + s) * 64 + dim] = (_Float16)((float)val * 0.125f);
    else if (part == 1)
        k[((size_t)bh * 2048 + s) * 64 + dim] = val;
    else
        vt[((size_t)bh * 64 + dim) * 2048 + s] = val;
}

// ---------------------------------------------------------------------------
// LayerNorm (f32 in -> f16 out), one 256-thread block per 1024-wide row.
// ---------------------------------------------------------------------------
__global__ void __launch_bounds__(256) layernorm_f16(const float* __restrict__ x,
                                                     const float* __restrict__ g,
                                                     const float* __restrict__ b,
                                                     _Float16* __restrict__ out) {
    __shared__ float red[2][8];
    int row = blockIdx.x, t = threadIdx.x;
    const float* xr = x + (size_t)row * 1024;
    v4f xv = *(const v4f*)(xr + t * 4);
    float s  = xv[0] + xv[1] + xv[2] + xv[3];
    float s2 = xv[0]*xv[0] + xv[1]*xv[1] + xv[2]*xv[2] + xv[3]*xv[3];
    #pragma unroll
    for (int m = 16; m >= 1; m >>= 1) {
        s  += __shfl_xor(s,  m, 32);
        s2 += __shfl_xor(s2, m, 32);
    }
    if ((t & 31) == 0) { red[0][t >> 5] = s; red[1][t >> 5] = s2; }
    __syncthreads();
    float tot = 0.f, tot2 = 0.f;
    #pragma unroll
    for (int i = 0; i < 8; ++i) { tot += red[0][i]; tot2 += red[1][i]; }
    float mu  = tot * (1.0f / 1024.0f);
    float var = tot2 * (1.0f / 1024.0f) - mu * mu;
    float rs  = rsqrtf(var + 1e-5f);
    v4h ov;
    #pragma unroll
    for (int i = 0; i < 4; ++i)
        ov[i] = (_Float16)((xv[i] - mu) * rs * g[t * 4 + i] + b[t * 4 + i]);
    *(v4h*)(out + (size_t)row * 1024 + t * 4) = ov;
}

// ---------------------------------------------------------------------------
// Tiled WMMA GEMM: C[M,N] = A[M,K] * Bt[N,K]^T  (A row-major, B pre-transposed
// N-major; both f16). 128x128 block, 8 wave32s, wave = 32x64 (2x4 WMMA tiles),
// K-step 32, double-buffered async LDS staging.
// EPI: 0 = f16 out; 1 = +f32 residual, f32 out; 2 = +bias,ReLU, f16 out;
//      3 = +bias,+f32 residual, f32 out.
// ---------------------------------------------------------------------------
template <int EPI>
__global__ void __launch_bounds__(256) gemm_f16(const _Float16* __restrict__ A,
                                                const _Float16* __restrict__ Bt,
                                                int M, int N, int K,
                                                const float* __restrict__ bias,
                                                const float* __restrict__ resid,
                                                float* __restrict__ Cf,
                                                _Float16* __restrict__ Ch) {
    __shared__ _Float16 As[2][128 * 40];   // [m][k], +8 pad
    __shared__ _Float16 Bs[2][128 * 40];   // [n][k], +8 pad
    const int t    = threadIdx.x;
    const int lane = t & 31;
    const int wave = t >> 5;
    const int wm   = wave & 3;             // 4 waves along M
    const int wn   = wave >> 2;            // 2 waves along N
    const int hl   = lane >> 4;
    const int ln   = lane & 15;
    const int n0   = blockIdx.x * 128;
    const int m0   = blockIdx.y * 128;

    v8f acc[2][4];
    const v8f vzero = {};
    #pragma unroll
    for (int i = 0; i < 2; ++i)
        #pragma unroll
        for (int j = 0; j < 4; ++j) acc[i][j] = vzero;

    const int srow = t >> 1, scol = (t & 1) * 16;   // 16 halfs per thread

    const _Float16* aRow = A  + (size_t)(m0 + srow) * K + scol;
    const _Float16* bRow = Bt + (size_t)(n0 + srow) * K + scol;

    // stage tile kc into buffer buf
    auto stage = [&](int kc, int buf) {
        _Float16* al = &As[buf][srow * 40 + scol];
        _Float16* bl = &Bs[buf][srow * 40 + scol];
        copy8(aRow + kc,     al);
        copy8(aRow + kc + 8, al + 8);
        copy8(bRow + kc,     bl);
        copy8(bRow + kc + 8, bl + 8);
    };

    const int nk = K >> 5;
    stage(0, 0);
    for (int it = 0; it < nk; ++it) {
        wait_async();
        __syncthreads();                 // tile[cur] ready; prev buffer free
        const int cur = it & 1;
        if (it + 1 < nk) stage((it + 1) << 5, cur ^ 1);

        const _Float16* Ab = As[cur];
        const _Float16* Bb = Bs[cur];
        v16h af[2];
        #pragma unroll
        for (int i = 0; i < 2; ++i) {
            int r = wm * 32 + i * 16 + ln;
            af[i] = frag_from(&Ab[r * 40 + hl * 8], &Ab[r * 40 + 16 + hl * 8]);
        }
        v16h bf[4];
        #pragma unroll
        for (int j = 0; j < 4; ++j) {
            int n = wn * 64 + j * 16 + ln;
            bf[j] = frag_from(&Bb[n * 40 + hl * 8], &Bb[n * 40 + 16 + hl * 8]);
        }
        #pragma unroll
        for (int i = 0; i < 2; ++i)
            #pragma unroll
            for (int j = 0; j < 4; ++j)
                acc[i][j] = wmma_f16(af[i], bf[j], acc[i][j]);
    }

    // ---- epilogue ----
    #pragma unroll
    for (int i = 0; i < 2; ++i) {
        int mrowbase = m0 + wm * 32 + i * 16 + hl * 8;
        #pragma unroll
        for (int j = 0; j < 4; ++j) {
            int col = n0 + wn * 64 + j * 16 + ln;
            float bv = (EPI == 2 || EPI == 3) ? bias[col] : 0.0f;
            #pragma unroll
            for (int r = 0; r < 8; ++r) {
                size_t idx = (size_t)(mrowbase + r) * N + col;
                float v = acc[i][j][r] + bv;
                if (EPI == 2) v = fmaxf(v, 0.0f);
                if (EPI == 1 || EPI == 3) v += resid[idx];
                if (EPI == 0 || EPI == 2) Ch[idx] = (_Float16)v;
                else                      Cf[idx] = v;
            }
        }
    }
}

// ---------------------------------------------------------------------------
// Flash attention: block = (b*h, 128 q rows), wave owns 16 q rows.
// q pre-scaled by 1/sqrt(64); v given transposed [bh][64][2048].
// Double-buffered async K/Vt staging. Output attn16 [b][s][h*64+d] f16.
// ---------------------------------------------------------------------------
__global__ void __launch_bounds__(256) attention(const _Float16* __restrict__ q,
                                                 const _Float16* __restrict__ k,
                                                 const _Float16* __restrict__ vt,
                                                 _Float16* __restrict__ o) {
    __shared__ _Float16 Ql[128 * 72];
    __shared__ _Float16 Kl[2][128 * 72];
    __shared__ _Float16 Vt[2][64 * 136];     // [dim][key]
    __shared__ _Float16 Pl[8 * 16 * 136];    // per-wave P tiles
    const int t    = threadIdx.x;
    const int lane = t & 31;
    const int wave = t >> 5;
    const int hl   = lane >> 4;
    const int ln   = lane & 15;
    const int bh   = blockIdx.y;
    const int qb   = blockIdx.x;
    const size_t baseQK = (size_t)bh * 2048 * 64;
    const size_t baseV  = (size_t)bh * 64 * 2048;

    const int krow = t >> 1, kd0 = (t & 1) * 32;   // K/Q staging: 32 halfs/thread
    const int vdim = t >> 2, vk0 = (t & 3) * 32;   // Vt staging: 32 halfs/thread

    auto stageKV = [&](int kb, int buf) {
        const _Float16* kg = k + baseQK + (size_t)(kb * 128 + krow) * 64 + kd0;
        _Float16* kl = &Kl[buf][krow * 72 + kd0];
        copy8(kg,      kl);      copy8(kg + 8,  kl + 8);
        copy8(kg + 16, kl + 16); copy8(kg + 24, kl + 24);
        const _Float16* vg = vt + baseV + (size_t)vdim * 2048 + kb * 128 + vk0;
        _Float16* vl = &Vt[buf][vdim * 136 + vk0];
        copy8(vg,      vl);      copy8(vg + 8,  vl + 8);
        copy8(vg + 16, vl + 16); copy8(vg + 24, vl + 24);
    };

    // stage Q block (128 x 64) + first K/V tile
    {
        const _Float16* g = q + baseQK + (size_t)(qb * 128 + krow) * 64 + kd0;
        _Float16* l = &Ql[krow * 72 + kd0];
        copy8(g,      l);      copy8(g + 8,  l + 8);
        copy8(g + 16, l + 16); copy8(g + 24, l + 24);
    }
    stageKV(0, 0);
    wait_async();
    __syncthreads();

    v16h qf[2];
    {
        int row = wave * 16 + ln;
        qf[0] = frag_from(&Ql[row * 72 + hl * 8],      &Ql[row * 72 + 16 + hl * 8]);
        qf[1] = frag_from(&Ql[row * 72 + 32 + hl * 8], &Ql[row * 72 + 48 + hl * 8]);
    }

    float mrow[8], lrow[8];
    v8f oacc[4];
    const v8f vzero = {};
    #pragma unroll
    for (int r = 0; r < 8; ++r) { mrow[r] = -1e30f; lrow[r] = 0.0f; }
    #pragma unroll
    for (int j = 0; j < 4; ++j) oacc[j] = vzero;

    for (int kb = 0; kb < 16; ++kb) {
        const int cur = kb & 1;
        if (kb + 1 < 16) stageKV(kb + 1, cur ^ 1);   // overlap with compute
        const _Float16* Kb = Kl[cur];
        const _Float16* Vb = Vt[cur];

        // scores S = Q Kt : 8 n-tiles x (K=64 in 2 chunks)
        v8f sacc[8];
        #pragma unroll
        for (int nt = 0; nt < 8; ++nt) sacc[nt] = vzero;
        #pragma unroll
        for (int nt = 0; nt < 8; ++nt) {
            int n = nt * 16 + ln;
            v16h kf0 = frag_from(&Kb[n * 72 + hl * 8],      &Kb[n * 72 + 16 + hl * 8]);
            v16h kf1 = frag_from(&Kb[n * 72 + 32 + hl * 8], &Kb[n * 72 + 48 + hl * 8]);
            sacc[nt] = wmma_f16(qf[0], kf0, sacc[nt]);
            sacc[nt] = wmma_f16(qf[1], kf1, sacc[nt]);
        }

        // online softmax
        float corr[8];
        #pragma unroll
        for (int r = 0; r < 8; ++r) {
            float mx = sacc[0][r];
            #pragma unroll
            for (int nt = 1; nt < 8; ++nt) mx = fmaxf(mx, sacc[nt][r]);
            mx = fmaxf(mx, __shfl_xor(mx, 1, 32));
            mx = fmaxf(mx, __shfl_xor(mx, 2, 32));
            mx = fmaxf(mx, __shfl_xor(mx, 4, 32));
            mx = fmaxf(mx, __shfl_xor(mx, 8, 32));
            float mnew = fmaxf(mrow[r], mx);
            float c = __builtin_exp2f((mrow[r] - mnew) * LOG2E);
            float rs = 0.0f;
            #pragma unroll
            for (int nt = 0; nt < 8; ++nt) {
                float p = __builtin_exp2f((sacc[nt][r] - mnew) * LOG2E);
                sacc[nt][r] = p;
                rs += p;
            }
            rs += __shfl_xor(rs, 1, 32);
            rs += __shfl_xor(rs, 2, 32);
            rs += __shfl_xor(rs, 4, 32);
            rs += __shfl_xor(rs, 8, 32);
            lrow[r] = lrow[r] * c + rs;
            mrow[r] = mnew;
            corr[r] = c;
        }
        #pragma unroll
        for (int j = 0; j < 4; ++j)
            #pragma unroll
            for (int r = 0; r < 8; ++r) oacc[j][r] *= corr[r];

        // P to wave-private LDS (same-wave DS ops stay in order; no barrier)
        _Float16* Pw = &Pl[wave * 16 * 136];
        #pragma unroll
        for (int nt = 0; nt < 8; ++nt)
            #pragma unroll
            for (int r = 0; r < 8; ++r)
                Pw[(r + 8 * hl) * 136 + nt * 16 + ln] = (_Float16)sacc[nt][r];

        // O += P * V : 128 keys in 4 chunks, 4 n-tiles (dim 64)
        #pragma unroll
        for (int kc2 = 0; kc2 < 128; kc2 += 32) {
            v16h pf = frag_from(&Pw[ln * 136 + kc2 + hl * 8],
                                &Pw[ln * 136 + kc2 + 16 + hl * 8]);
            #pragma unroll
            for (int j = 0; j < 4; ++j) {
                int d = j * 16 + ln;
                v16h vf = frag_from(&Vb[d * 136 + kc2 + hl * 8],
                                    &Vb[d * 136 + kc2 + 16 + hl * 8]);
                oacc[j] = wmma_f16(pf, vf, oacc[j]);
            }
        }
        wait_async();
        __syncthreads();                 // next tile ready; cur free to rewrite
    }

    // write O / l  ->  attn16 [b][s][h*64+d]
    int b = bh >> 4, h = bh & 15;
    #pragma unroll
    for (int r = 0; r < 8; ++r) {
        int s = qb * 128 + wave * 16 + r + 8 * hl;
        float inv = 1.0f / lrow[r];
        #pragma unroll
        for (int j = 0; j < 4; ++j) {
            int d = j * 16 + ln;
            o[((size_t)(b * 2048 + s)) * 1024 + h * 64 + d] =
                (_Float16)(oacc[j][r] * inv);
        }
    }
}

// ---------------------------------------------------------------------------
// host side
// ---------------------------------------------------------------------------
extern "C" void kernel_launch(void* const* d_in, const int* in_sizes, int n_in,
                              void* d_out, int out_size, void* d_ws, size_t ws_size,
                              hipStream_t stream) {
    const float* x     = (const float*)d_in[0];
    const float* W_Q   = (const float*)d_in[1];
    const float* W_K   = (const float*)d_in[2];
    const float* W_V   = (const float*)d_in[3];
    const float* W_O   = (const float*)d_in[4];
    const float* ln1g  = (const float*)d_in[5];
    const float* ln1b  = (const float*)d_in[6];
    const float* ln2g  = (const float*)d_in[7];
    const float* ln2b  = (const float*)d_in[8];
    const float* w1    = (const float*)d_in[9];
    const float* b1    = (const float*)d_in[10];
    const float* w2    = (const float*)d_in[11];
    const float* b2    = (const float*)d_in[12];
    float* out = (float*)d_out;

    char* ws = (char*)d_ws;
    size_t off = 0;
    auto take = [&](size_t bytes) -> char* {
        off = (off + 255) & ~(size_t)255;
        char* p = ws + off;
        off += bytes;
        return p;
    };
    _Float16* wqkvT  = (_Float16*)take((size_t)3072 * 1024 * 2);  // [3072][1024]
    _Float16* woT    = (_Float16*)take((size_t)1024 * 1024 * 2);  // [1024][1024] (n-major)
    _Float16* w1T    = (_Float16*)take((size_t)4096 * 1024 * 2);  // [4096][1024]
    _Float16* w2T    = (_Float16*)take((size_t)1024 * 4096 * 2);  // [1024][4096]
    _Float16* ln16   = (_Float16*)take((size_t)4096 * 1024 * 2);  // reused ln1/ln2
    _Float16* qkv16  = (_Float16*)take((size_t)4096 * 3072 * 2);
    _Float16* q16    = (_Float16*)take((size_t)2 * 16 * 2048 * 64 * 2);
    _Float16* k16    = (_Float16*)take((size_t)2 * 16 * 2048 * 64 * 2);
    _Float16* v16t   = (_Float16*)take((size_t)2 * 16 * 64 * 2048 * 2);
    _Float16* attn16 = (_Float16*)take((size_t)4096 * 1024 * 2);
    float*    x2     = (float*)   take((size_t)4096 * 1024 * 4);
    _Float16* m16    = (_Float16*)take((size_t)4096 * 4096 * 2);

    // weight conversions (all pre-transposed to N-major for async staging)
    build_wqkv_t<<<(3072 * 1024) / 256, 256, 0, stream>>>(W_Q, W_K, W_V, wqkvT);
    cvt_f32_to_f16_t<<<(1024 * 1024) / 256, 256, 0, stream>>>(W_O, woT, 1024, 1024);
    cvt_f32_to_f16_t<<<(1024 * 4096) / 256, 256, 0, stream>>>(w1, w1T, 1024, 4096);
    cvt_f32_to_f16_t<<<(4096 * 1024) / 256, 256, 0, stream>>>(w2, w2T, 4096, 1024);

    // LN1 -> f16
    layernorm_f16<<<4096, 256, 0, stream>>>(x, ln1g, ln1b, ln16);

    // QKV projection: [4096,1024] x [1024,3072]
    gemm_f16<0><<<dim3(24, 32), 256, 0, stream>>>(ln16, wqkvT, 4096, 3072, 1024,
                                                  nullptr, nullptr, nullptr, qkv16);
    scatter_qkv<<<(4096 * 3072) / 256, 256, 0, stream>>>(qkv16, q16, k16, v16t);

    // attention: grid (q blocks of 128, b*h)
    attention<<<dim3(16, 32), 256, 0, stream>>>(q16, k16, v16t, attn16);

    // O projection + residual: x2 = x + attn @ W_O
    gemm_f16<1><<<dim3(8, 32), 256, 0, stream>>>(attn16, woT, 4096, 1024, 1024,
                                                 nullptr, x, x2, nullptr);

    // LN2 -> f16
    layernorm_f16<<<4096, 256, 0, stream>>>(x2, ln2g, ln2b, ln16);

    // MLP1: relu(h @ w1 + b1)
    gemm_f16<2><<<dim3(32, 32), 256, 0, stream>>>(ln16, w1T, 4096, 4096, 1024,
                                                  b1, nullptr, nullptr, m16);

    // MLP2 + residual: out = x2 + m @ w2 + b2
    gemm_f16<3><<<dim3(8, 32), 256, 0, stream>>>(m16, w2T, 4096, 1024, 4096,
                                                 b2, x2, out, nullptr);
}
